// ODEBlock_27092653703612
// MI455X (gfx1250) — compile-verified, matrix operand order checked by reading, not run on hardware
//
#include <hip/hip_runtime.h>

typedef unsigned short u16;
typedef __attribute__((ext_vector_type(16))) __bf16 v16bf;
typedef __attribute__((ext_vector_type(8)))  float  v8f;
typedef __attribute__((ext_vector_type(4)))  unsigned int u32x4;
typedef __attribute__((ext_vector_type(8)))  int i32x8;
typedef __attribute__((ext_vector_type(4)))  int i32x4;

// float -> bf16 round-to-nearest-even
__device__ __forceinline__ u16 f2bf(float x) {
    unsigned int u = __float_as_uint(x);
    u += 0x7FFFu + ((u >> 16) & 1u);
    return (u16)(u >> 16);
}

struct Frag32 { uint4 lo, hi; };   // 16 bf16 elements

__device__ __forceinline__ v16bf frag_cast(const Frag32& f) {
    v16bf v;
    __builtin_memcpy(&v, &f, sizeof(v));
    return v;
}

// -------------------------------------------------------------------------
// TDM issue: DMA a 2D bf16 tile [tileRows x 32] from a row-major matrix
// (row length = rowLen elements) into LDS at ldsByteAddr, inserting 16B of
// LDS padding every 64B (row pitch 80B -> matches PAD=40 fragment indexing).
// Descriptor packing per CDNA5 ISA §8.3/§8.4:
//   group0: count=1 | lds_addr | global_addr[56:0] | type=2
//   group1: data_size=2B, pad_enable, pad_interval=16DW, pad_amount=4DW,
//           tensor_dim0=rowLen, tensor_dim1=totRows, tile_dim0=32,
//           tile_dim1=tileRows, tensor_dim0_stride=rowLen
// -------------------------------------------------------------------------
__device__ __forceinline__ void tdm_issue_tile(const u16* tileGlobal,
                                               unsigned ldsByteAddr,
                                               int rowLen, int totRows,
                                               int tileRows)
{
    unsigned long long ga = (unsigned long long)(uintptr_t)tileGlobal;
    u32x4 g0;
    g0.x = 1u;                                                  // count=1, user desc
    g0.y = ldsByteAddr;                                         // lds_addr
    g0.z = (unsigned)ga;                                        // global_addr[31:0]
    g0.w = ((unsigned)(ga >> 32) & 0x01FFFFFFu) | 0x80000000u;  // addr[56:32] | type=2

    i32x8 g1;
    g1[0] = (int)((1u << 16)      // data_size = 2 bytes
                | (1u << 20)      // pad_enable
                | (3u << 22)      // pad_interval: 16 DWORDs (64B)
                | (3u << 25));    // pad_amount:   4 DWORDs (16B)
    g1[1] = (int)(((unsigned)rowLen & 0xFFFFu) << 16);                       // tensor_dim0 lo
    g1[2] = (int)((((unsigned)rowLen) >> 16) | (((unsigned)totRows) << 16)); // dim0 hi | dim1 lo
    g1[3] = (int)((((unsigned)totRows) >> 16) | (32u << 16));                // dim1 hi | tile_dim0=32
    g1[4] = tileRows;                                                        // tile_dim1 (tile_dim2=0)
    g1[5] = rowLen;                                                          // tensor_dim0_stride lo
    g1[6] = 0;
    g1[7] = 0;

    i32x4 gz = {0, 0, 0, 0};
#if __has_include(<hip/amd_detail/amd_gfx1250_TDM.h>)
    i32x8 z8 = {};
    __builtin_amdgcn_tensor_load_to_lds(g0, g1, gz, gz, z8, 0);
#else
    __builtin_amdgcn_tensor_load_to_lds(g0, g1, gz, gz, 0);
#endif
}

// -------------------------------------------------------------------------
// Tiled bf16 WMMA GEMM:  out = act(A[MxK] * Bt[NxK]^T + bias)
//   A  : bf16 row-major [M][K];  Bt : bf16 row-major [N][K] (K contiguous)
//   ACT==1 : out = bf16(tanh(v));  ACT==0 : out = f32(v)
// 256 threads = 8 waves (4x2). Block tile 256x128, BK=32.
// Wave tile 64x64 -> 4x4 v_wmma_f32_16x16x32_bf16 per K-step.
// Operand staging: double-buffered TDM (tensor_load_to_lds), TENSORcnt-gated.
// -------------------------------------------------------------------------
template<int ACT>
__global__ __launch_bounds__(256)
void gemm_bf16_wmma(const u16* __restrict__ A, const u16* __restrict__ Bt,
                    const float* __restrict__ bias, void* __restrict__ outp,
                    int M, int N, int K)
{
    constexpr int BM = 256, BN = 128, BK = 32, PAD = 40;  // 80B LDS row pitch
    __shared__ __align__(16) u16 lA[2][BM * PAD];
    __shared__ __align__(16) u16 lB[2][BN * PAD];

    const int tid   = threadIdx.x;
    const int lane  = tid & 31;
    const int wid   = tid >> 5;
    const int waveM = wid >> 1;          // 0..3 -> 64-row slab
    const int waveN = wid & 1;           // 0..1 -> 64-col slab
    const int blockM = blockIdx.y * BM;
    const int blockN = blockIdx.x * BN;

    v8f acc[4][4] = {};

    // prologue: DMA first K-tile into buffer 0 (wave 0 issues; EXEC-independent)
    if (wid == 0) {
        tdm_issue_tile(A  + (size_t)blockM * K, (unsigned)(uintptr_t)&lA[0][0], K, M, BM);
        tdm_issue_tile(Bt + (size_t)blockN * K, (unsigned)(uintptr_t)&lB[0][0], K, N, BN);
    }

    const int nk = K / BK;
    for (int it = 0; it < nk; ++it) {
        const int  cur     = it & 1;
        const bool hasNext = (it + 1) < nk;

        if (wid == 0 && hasNext) {   // prefetch next K-tile into the other buffer
            const int kn = (it + 1) * BK;
            tdm_issue_tile(A  + (size_t)blockM * K + kn,
                           (unsigned)(uintptr_t)&lA[cur ^ 1][0], K, M, BM);
            tdm_issue_tile(Bt + (size_t)blockN * K + kn,
                           (unsigned)(uintptr_t)&lB[cur ^ 1][0], K, N, BN);
        }
        // TDM completes in-order per wave: <=2 outstanding leaves current tile done
        if (hasNext) __builtin_amdgcn_s_wait_tensorcnt(2);
        else         __builtin_amdgcn_s_wait_tensorcnt(0);
        __syncthreads();

        // fragment loads per ISA 16-bit A layout + 16 WMMAs
        {
            const int kb = (lane >> 4) * 8;  // lanes 0-15: K 0-7/16-23, 16-31: K 8-15/24-31
            const int lr = lane & 15;
            v16bf bfr[4];
            #pragma unroll
            for (int nt = 0; nt < 4; ++nt) {
                int nn = 64 * waveN + 16 * nt + lr;
                Frag32 f;
                f.lo = *(const uint4*)(&lB[cur][nn * PAD + kb]);
                f.hi = *(const uint4*)(&lB[cur][nn * PAD + 16 + kb]);
                bfr[nt] = frag_cast(f);
            }
            #pragma unroll
            for (int mt = 0; mt < 4; ++mt) {
                int mm = 64 * waveM + 16 * mt + lr;
                Frag32 f;
                f.lo = *(const uint4*)(&lA[cur][mm * PAD + kb]);
                f.hi = *(const uint4*)(&lA[cur][mm * PAD + 16 + kb]);
                v16bf afr = frag_cast(f);
                #pragma unroll
                for (int nt = 0; nt < 4; ++nt) {
                    acc[mt][nt] = __builtin_amdgcn_wmma_f32_16x16x32_bf16(
                        false, afr, false, bfr[nt], (short)0, acc[mt][nt], false, false);
                }
            }
        }
        __syncthreads();   // all waves done reading before this buffer is re-targeted
    }

    // epilogue: C/D layout -> lane (l&15)=N, VGPR r: M = r + 8*(l>>4)
    const int lr  = lane & 15;
    const int lhi = lane >> 4;
    #pragma unroll
    for (int mt = 0; mt < 4; ++mt) {
        #pragma unroll
        for (int nt = 0; nt < 4; ++nt) {
            #pragma unroll
            for (int r = 0; r < 8; ++r) {
                int gm = blockM + 64 * waveM + 16 * mt + 8 * lhi + r;
                int gn = blockN + 64 * waveN + 16 * nt + lr;
                float v = acc[mt][nt][r] + bias[gn];
                if (ACT) {
                    ((u16*)outp)[(size_t)gm * N + gn] = f2bf(tanhf(v));
                } else {
                    ((float*)outp)[(size_t)gm * N + gn] = v;
                }
            }
        }
    }
}

// ---- transpose + f32->bf16 convert (weights become K-contiguous "Bt") ----
__global__ void transpose_cvt_bf16(const float* __restrict__ in, u16* __restrict__ out,
                                   int R, int C)
{
    __shared__ u16 tile[32][33];
    int tx = threadIdx.x, ty = threadIdx.y;     // 32 x 8
    int r0 = blockIdx.y * 32, c0 = blockIdx.x * 32;
    #pragma unroll
    for (int i = 0; i < 4; ++i)
        tile[ty + 8 * i][tx] = f2bf(in[(size_t)(r0 + ty + 8 * i) * C + c0 + tx]);
    __syncthreads();
    #pragma unroll
    for (int i = 0; i < 4; ++i)
        out[(size_t)(c0 + ty + 8 * i) * R + r0 + tx] = tile[tx][ty + 8 * i];
}

// ---- y = x ; ybf = bf16(x) ----
__global__ void node_init(const float* __restrict__ x, float* __restrict__ y,
                          u16* __restrict__ ybf, int n)
{
    int i = blockIdx.x * blockDim.x + threadIdx.x;
    if (i < n) { float v = x[i]; y[i] = v; ybf[i] = f2bf(v); }
}

// ---- fused RK4 stage update ----
// mode 0: acc = y + cAcc*k ; ybf = bf16(y + cY*k)         (after k1)
// mode 1: acc += cAcc*k    ; ybf = bf16(y + cY*k)         (after k2,k3)
// mode 2: y = acc + cAcc*k ; ybf = bf16(y)                (after k4)
__global__ void rk_update(float* __restrict__ y, float* __restrict__ acc,
                          const float* __restrict__ k, u16* __restrict__ ybf,
                          float cAcc, float cY, int mode, int n)
{
    int i = blockIdx.x * blockDim.x + threadIdx.x;
    if (i >= n) return;
    float kv = k[i];
    if (mode == 0) {
        float yv = y[i];
        acc[i] = fmaf(cAcc, kv, yv);
        ybf[i] = f2bf(fmaf(cY, kv, yv));
    } else if (mode == 1) {
        acc[i] = fmaf(cAcc, kv, acc[i]);
        ybf[i] = f2bf(fmaf(cY, kv, y[i]));
    } else {
        float yn = fmaf(cAcc, kv, acc[i]);
        y[i] = yn;
        ybf[i] = f2bf(yn);
    }
}

extern "C" void kernel_launch(void* const* d_in, const int* in_sizes, int n_in,
                              void* d_out, int out_size, void* d_ws, size_t ws_size,
                              hipStream_t stream)
{
    (void)in_sizes; (void)n_in; (void)out_size; (void)ws_size;
    const int B = 4096, D = 1024, H = 4096;

    const float* x  = (const float*)d_in[0];
    const float* W1 = (const float*)d_in[1];  // [D][H]
    const float* b1 = (const float*)d_in[2];  // [H]
    const float* W2 = (const float*)d_in[3];  // [H][D]
    const float* b2 = (const float*)d_in[4];  // [D]

    // workspace carve-up (88 MB total)
    char* ws = (char*)d_ws;
    u16*   W1t  = (u16*)ws;   ws += (size_t)H * D * sizeof(u16);    // bf16 [H][D]
    u16*   W2t  = (u16*)ws;   ws += (size_t)D * H * sizeof(u16);    // bf16 [D][H]
    u16*   Ybf  = (u16*)ws;   ws += (size_t)B * D * sizeof(u16);    // bf16 eval input
    u16*   Hbf  = (u16*)ws;   ws += (size_t)B * H * sizeof(u16);    // bf16 hidden
    float* accb = (float*)ws; ws += (size_t)B * D * sizeof(float);  // RK accumulator
    float* kbuf = (float*)ws; ws += (size_t)B * D * sizeof(float);  // stage derivative
    float* y    = (float*)d_out;                                    // f32 state lives in d_out

    // weights -> bf16, K-contiguous
    dim3 tblk(32, 8);
    transpose_cvt_bf16<<<dim3(H / 32, D / 32), tblk, 0, stream>>>(W1, W1t, D, H);
    transpose_cvt_bf16<<<dim3(D / 32, H / 32), tblk, 0, stream>>>(W2, W2t, H, D);

    const int n = B * D;
    node_init<<<n / 256, 256, 0, stream>>>(x, y, Ybf, n);

    const int NSTEPS = 8;                 // fixed-step RK4, global err O(h^4) << 1e-3
    const float h = 1.0f / (float)NSTEPS;
    dim3 g1(H / 128, B / 256);            // GEMM1 out: [B][H]
    dim3 g2(D / 128, B / 256);            // GEMM2 out: [B][D]

    for (int s = 0; s < NSTEPS; ++s) {
        for (int st = 0; st < 4; ++st) {
            // k = f(ystage):  H = tanh(Y*W1 + b1) ; k = H*W2 + b2
            gemm_bf16_wmma<1><<<g1, 256, 0, stream>>>(Ybf, W1t, b1, (void*)Hbf,  B, H, D);
            gemm_bf16_wmma<0><<<g2, 256, 0, stream>>>(Hbf, W2t, b2, (void*)kbuf, B, D, H);
            if (st == 0)
                rk_update<<<n / 256, 256, 0, stream>>>(y, accb, kbuf, Ybf, h / 6.f, h / 2.f, 0, n);
            else if (st == 1)
                rk_update<<<n / 256, 256, 0, stream>>>(y, accb, kbuf, Ybf, h / 3.f, h / 2.f, 1, n);
            else if (st == 2)
                rk_update<<<n / 256, 256, 0, stream>>>(y, accb, kbuf, Ybf, h / 3.f, h,       1, n);
            else
                rk_update<<<n / 256, 256, 0, stream>>>(y, accb, kbuf, Ybf, h / 6.f, 0.f,     2, n);
        }
    }
    // final state (t = 1.0) is already in d_out
}
